// CustomLSTMCell_31911607009503
// MI455X (gfx1250) — matrix operand order repaired and checked by reading, not compiled, not run on hardware
//
#include <hip/hip_runtime.h>
#include <hip/hip_bf16.h>

typedef __attribute__((ext_vector_type(16))) __bf16 v16bf;
typedef __attribute__((ext_vector_type(8)))  __bf16 v8bf;
typedef __attribute__((ext_vector_type(8)))  float  v8f;
typedef __attribute__((ext_vector_type(4)))  int    v4i;

#define IN_SZ  2048
#define HID    2048
#define BATCH  4096
#define KDIM   (IN_SZ + HID)   // 4096
#define NGATE  4

#define BM  128
#define BN  64
#define BK  32
#define LDK 40                 // padded K stride in bf16 elems (80B = 20 banks, conflict-free)
#define NKT (KDIM / BK)        // 128 K-steps

// ---- CDNA5 async global->LDS copy (no data VGPRs, tracked by ASYNCcnt) ----
#if __has_builtin(__builtin_amdgcn_global_load_async_to_lds_b128) && \
    __has_builtin(__builtin_amdgcn_s_wait_asynccnt)
#define USE_ASYNC_LDS 1
#else
#define USE_ASYNC_LDS 0
#endif

__device__ __forceinline__ void copy16_async(const __bf16* g, __bf16* l) {
#if USE_ASYNC_LDS
  __builtin_amdgcn_global_load_async_to_lds_b128((v4i*)g, (v4i*)l, 0, 0);
#else
  *(uint4*)l = *(const uint4*)g;
#endif
}

__device__ __forceinline__ void wait_async_done() {
#if USE_ASYNC_LDS
  __builtin_amdgcn_s_wait_asynccnt(0);
#endif
}

// ---- short transcendental forms: one v_exp_f32 each ----
__device__ __forceinline__ float sigmoidf_(float x) {
  return __frcp_rn(1.0f + __expf(-x));
}
__device__ __forceinline__ float tanhf_(float x) {
  // tanh(x) = 2*sigmoid(2x) - 1
  return __builtin_fmaf(2.0f, __frcp_rn(1.0f + __expf(-2.0f * x)), -1.0f);
}

// ---------------------------------------------------------------------------
// Kernel 1: fuse concat([x, h]) and f32 -> bf16 convert into A_bf [BATCH][KDIM]
// ---------------------------------------------------------------------------
__global__ void pack_a(const float* __restrict__ x, const float* __restrict__ h,
                       __bf16* __restrict__ a) {
  int e = (blockIdx.x * blockDim.x + threadIdx.x) * 4;   // 4 floats per thread
  int r = e >> 12;           // / KDIM
  int c = e & (KDIM - 1);
  const float* src = (c < IN_SZ) ? (x + (size_t)r * IN_SZ + c)
                                 : (h + (size_t)r * HID + (c - IN_SZ));
  float4 v = *(const float4*)src;
  __bf16* dst = a + (size_t)e;
  dst[0] = (__bf16)v.x; dst[1] = (__bf16)v.y;
  dst[2] = (__bf16)v.z; dst[3] = (__bf16)v.w;
}

// ---------------------------------------------------------------------------
// Kernel 2: LDS-tiled transpose + convert: Wt[g][n][k] = bf16(W_g[k][n])
// ---------------------------------------------------------------------------
__global__ void conv_w(const float* __restrict__ Wf, const float* __restrict__ Wi,
                       const float* __restrict__ Wc, const float* __restrict__ Wo,
                       __bf16* __restrict__ Wt) {
  __shared__ float tile[32][33];
  const int g = blockIdx.z;
  const float* W = (g == 0) ? Wf : (g == 1) ? Wi : (g == 2) ? Wc : Wo;
  const int k0 = blockIdx.y * 32;
  const int n0 = blockIdx.x * 32;
  const int tx = threadIdx.x & 31;
  const int ty = threadIdx.x >> 5;          // 32x8 threads
#pragma unroll
  for (int i = 0; i < 32; i += 8)
    tile[ty + i][tx] = W[(size_t)(k0 + ty + i) * HID + n0 + tx];
  __syncthreads();
  __bf16* dst = Wt + ((size_t)g * HID + n0) * KDIM + k0;
#pragma unroll
  for (int i = 0; i < 32; i += 8)
    dst[(size_t)(ty + i) * KDIM + tx] = (__bf16)tile[tx][ty + i];
}

// ---------------------------------------------------------------------------
// Kernel 3: fused 4-gate GEMM + LSTM epilogue.
//   Block tile: 128(M) x 64(N), 8 waves (wave32), wave tile 32x32 per gate.
//   K pipeline: async global->LDS DMA for stage kt+1 overlapped with the
//   16 WMMAs of stage kt; s_wait_asynccnt + one barrier per step.
// ---------------------------------------------------------------------------
__global__ __launch_bounds__(256) void lstm_gemm(
    const __bf16* __restrict__ A,    // [BATCH][KDIM] bf16
    const __bf16* __restrict__ Wt,   // [NGATE][HID][KDIM] bf16 (pre-transposed)
    const float* __restrict__ bfv, const float* __restrict__ biv,
    const float* __restrict__ bcv, const float* __restrict__ bov,
    const float* __restrict__ cprev,
    float* __restrict__ hout, float* __restrict__ cout) {
  __shared__ __bf16 As[2][BM * LDK];          // 2 * 10240 B
  __shared__ __bf16 Bs[2][NGATE][BN * LDK];   // 2 * 20480 B  (total 60 KB)

  const int tid  = threadIdx.x;
  const int lane = tid & 31;
  const int wid  = tid >> 5;
  const int wm   = wid & 3;        // 4 waves along M
  const int wn   = wid >> 2;       // 2 waves along N
  const int m0   = blockIdx.y * BM;
  const int n0   = blockIdx.x * BN;
  const int half = lane >> 4;      // K-half selector (lane half holds contiguous K 16*half..+15)
  const int l16  = lane & 15;

  const v8f vzero = {0.f, 0.f, 0.f, 0.f, 0.f, 0.f, 0.f, 0.f};
  v8f acc[NGATE][2][2];
#pragma unroll
  for (int g = 0; g < NGATE; ++g)
#pragma unroll
    for (int mi = 0; mi < 2; ++mi)
#pragma unroll
      for (int ni = 0; ni < 2; ++ni) acc[g][mi][ni] = vzero;

  // ---- per-thread copy addresses (advance global ptrs by BK each stage) ----
  const int aRow0 = tid >> 2;                 // rows 0..63
  const int aRow1 = (tid + 256) >> 2;         // rows 64..127
  const int kc    = (tid & 3) << 3;           // 0/8/16/24 (bf16 elems)
  const int bn    = tid >> 2;                 // 0..63

  const __bf16* gA0 = A + (size_t)(m0 + aRow0) * KDIM + kc;
  const __bf16* gA1 = A + (size_t)(m0 + aRow1) * KDIM + kc;
  const __bf16* gB0 = Wt + ((size_t)0 * HID + n0 + bn) * KDIM + kc;
  const __bf16* gB1 = Wt + ((size_t)1 * HID + n0 + bn) * KDIM + kc;
  const __bf16* gB2 = Wt + ((size_t)2 * HID + n0 + bn) * KDIM + kc;
  const __bf16* gB3 = Wt + ((size_t)3 * HID + n0 + bn) * KDIM + kc;

  const int aOff0 = aRow0 * LDK + kc;
  const int aOff1 = aRow1 * LDK + kc;
  const int bOff  = bn * LDK + kc;

  auto issueStage = [&](int kt, int buf) {
    const int ko = kt * BK;
    copy16_async(gA0 + ko, &As[buf][aOff0]);
    copy16_async(gA1 + ko, &As[buf][aOff1]);
    copy16_async(gB0 + ko, &Bs[buf][0][bOff]);
    copy16_async(gB1 + ko, &Bs[buf][1][bOff]);
    copy16_async(gB2 + ko, &Bs[buf][2][bOff]);
    copy16_async(gB3 + ko, &Bs[buf][3][bOff]);
  };

  auto computeStage = [&](int buf) {
    v16bf af[2];
#pragma unroll
    for (int mi = 0; mi < 2; ++mi) {
      // A 16x32 bf16 layout: lanes<16 hold K 0..15, lanes>=16 hold K 16..31
      const __bf16* ap = &As[buf][(wm * 32 + mi * 16 + l16) * LDK];
      v8bf alo = *(const v8bf*)(ap + half * 16);
      v8bf ahi = *(const v8bf*)(ap + half * 16 + 8);
      af[mi] = __builtin_shufflevector(alo, ahi, 0, 1, 2, 3, 4, 5, 6, 7,
                                       8, 9, 10, 11, 12, 13, 14, 15);
    }
#pragma unroll
    for (int g = 0; g < NGATE; ++g) {
      // B 32x16 bf16 layout: lane holds 16 contiguous K of column N=l16,
      // lanes<16 -> K 0..15, lanes>=16 -> K 16..31
      const __bf16* bp0 = &Bs[buf][g][(wn * 32 + l16) * LDK + half * 16];
      const __bf16* bp1 = bp0 + 16 * LDK;
      v8bf b00 = *(const v8bf*)bp0;
      v8bf b01 = *(const v8bf*)(bp0 + 8);
      v8bf b10 = *(const v8bf*)bp1;
      v8bf b11 = *(const v8bf*)(bp1 + 8);
      v16bf bf0 = __builtin_shufflevector(b00, b01, 0, 1, 2, 3, 4, 5, 6, 7,
                                          8, 9, 10, 11, 12, 13, 14, 15);
      v16bf bf1 = __builtin_shufflevector(b10, b11, 0, 1, 2, 3, 4, 5, 6, 7,
                                          8, 9, 10, 11, 12, 13, 14, 15);
      acc[g][0][0] = __builtin_amdgcn_wmma_f32_16x16x32_bf16(
          false, af[0], false, bf0, (short)0, acc[g][0][0], false, false);
      acc[g][1][0] = __builtin_amdgcn_wmma_f32_16x16x32_bf16(
          false, af[1], false, bf0, (short)0, acc[g][1][0], false, false);
      acc[g][0][1] = __builtin_amdgcn_wmma_f32_16x16x32_bf16(
          false, af[0], false, bf1, (short)0, acc[g][0][1], false, false);
      acc[g][1][1] = __builtin_amdgcn_wmma_f32_16x16x32_bf16(
          false, af[1], false, bf1, (short)0, acc[g][1][1], false, false);
    }
  };

  issueStage(0, 0);
  wait_async_done();
  __syncthreads();
#pragma unroll 1
  for (int kt = 0; kt < NKT; ++kt) {
    if (kt + 1 < NKT) issueStage(kt + 1, (kt + 1) & 1);
    computeStage(kt & 1);
    wait_async_done();   // our wave's async DMAs into buf[next] have landed
    __syncthreads();     // everyone done reading buf[cur] / writing buf[next]
  }

  // ---- fused LSTM epilogue: gates -> (h_t, c_t), no intermediate buffer ----
  // C/D layout: VGPR r, lanes 0-15 = (M=r, N=lane); lanes 16-31 = (M=r+8, N=lane-16)
#pragma unroll
  for (int ni = 0; ni < 2; ++ni) {
    const int col = n0 + wn * 32 + ni * 16 + l16;
    const float bF = bfv[col], bI = biv[col], bC = bcv[col], bO = bov[col];
#pragma unroll
    for (int mi = 0; mi < 2; ++mi) {
#pragma unroll
      for (int r = 0; r < 8; ++r) {
        const int row = m0 + wm * 32 + mi * 16 + r + half * 8;
        const size_t idx = (size_t)row * HID + col;
        const float fp = acc[0][mi][ni][r] + bF;
        const float ip = acc[1][mi][ni][r] + bI;
        const float cp = acc[2][mi][ni][r] + bC;
        const float op = acc[3][mi][ni][r] + bO;
        const float ft = sigmoidf_(fp);
        const float it = sigmoidf_(ip);
        const float ch = tanhf_(cp);
        const float ot = sigmoidf_(op);
        const float ct = ft * cprev[idx] + it * ch;
        hout[idx] = ot * tanhf_(ct);
        cout[idx] = ct;
      }
    }
  }
}

// ---------------------------------------------------------------------------
extern "C" void kernel_launch(void* const* d_in, const int* in_sizes, int n_in,
                              void* d_out, int out_size, void* d_ws, size_t ws_size,
                              hipStream_t stream) {
  const float* x  = (const float*)d_in[0];
  const float* h  = (const float*)d_in[1];
  const float* c  = (const float*)d_in[2];
  const float* Wf = (const float*)d_in[3];
  const float* bf_ = (const float*)d_in[4];
  const float* Wi = (const float*)d_in[5];
  const float* bi_ = (const float*)d_in[6];
  const float* Wc = (const float*)d_in[7];
  const float* bc_ = (const float*)d_in[8];
  const float* Wo = (const float*)d_in[9];
  const float* bo_ = (const float*)d_in[10];

  __bf16* Abf = (__bf16*)d_ws;                              // 32 MB
  __bf16* Wt  = Abf + (size_t)BATCH * KDIM;                 // 64 MB

  pack_a<<<(BATCH * KDIM / 4) / 256, 256, 0, stream>>>(x, h, Abf);

  dim3 tg(HID / 32, KDIM / 32, NGATE);
  conv_w<<<tg, 256, 0, stream>>>(Wf, Wi, Wc, Wo, Wt);

  float* hout = (float*)d_out;
  float* cout = hout + (size_t)BATCH * HID;
  dim3 gg(HID / BN, BATCH / BM);
  lstm_gemm<<<gg, 256, 0, stream>>>(Abf, Wt, bf_, bi_, bc_, bo_, c, hout, cout);
}
